// Encoder_45260365365327
// MI455X (gfx1250) — compile-verified, matrix operand order checked by reading, not run on hardware
//
#include <hip/hip_runtime.h>
#include <hip/hip_bf16.h>
#include <stdint.h>

// ---------------------------------------------------------------------------
// 3-layer GRU encoder for MI455X (gfx1250, wave32, WMMA bf16)
//   B=32, T=1000, D=80, H=512, L=3, reset_after=True
//
// Pipeline:
//  1) kprep  : pack weights to bf16 "fragment-linear" layout with gate-column
//              permutation (each recurrent block owns z/r/xh/hh slices).
//  2) kproj  : xp = tanh(x @ proj_W + b) -> bf16 A-fragments  (WMMA)
//  3) kgx0   : gx0 = xp @ K0 + b_in0 for all (b,t)            (WMMA, bulk)
//  4) krecur : persistent 16-block kernel, 1000 steps, 3 layers/step,
//              2 global barriers/step, double-buffered bf16 h-state
//              fragments, f32 h carried for accuracy.
// ---------------------------------------------------------------------------

typedef __attribute__((ext_vector_type(16))) __bf16        v16bf;
typedef __attribute__((ext_vector_type(8)))  float         v8f;
typedef __attribute__((ext_vector_type(8)))  unsigned int  v8u;

#define NBLK 16

__device__ __forceinline__ unsigned short f2bf(float f) {
  unsigned u = __float_as_uint(f);
  u += 0x7fffu + ((u >> 16) & 1u);              // round-to-nearest-even
  return (unsigned short)(u >> 16);
}
__device__ __forceinline__ float bf2f(unsigned short h) {
  return __uint_as_float(((unsigned)h) << 16);
}
__device__ __forceinline__ float sigm(float x) { return 1.0f / (1.0f + __expf(-x)); }
// Branch-free tanh: avoids ocml's exec-mask branches on the serial GRU path.
__device__ __forceinline__ float ftanh(float x) {
  float xc = fminf(fmaxf(x, -15.0f), 15.0f);
  float e  = __expf(xc + xc);                   // e^{2x}, finite after clamp
  return 1.0f - 2.0f / (e + 1.0f);
}

// Fragment slot (lane*16 + halfword index) for 16-bit WMMA operands.
// A 16x32 tile: lane = h*16 + row, VGPR/k mapping per CDNA5 ISA 7.12.2.
// B 32x16 tile: mirrored with lane = h*16 + col.
__device__ __forceinline__ int frag_slot(int ln, int k) {
  int h = (k >> 3) & 1;
  int v = ((k >> 4) << 2) + ((k & 7) >> 1);
  int p = k & 1;
  return ((h << 4) + ln) * 16 + (v * 2 + p);
}
__device__ __forceinline__ size_t afrag_off(int m, int k, int KT) {
  return (size_t)((m >> 4) * KT + (k >> 5)) * 512 + frag_slot(m & 15, k & 31);
}
__device__ __forceinline__ size_t bfrag_off(int k, int n, int KT) {
  return (size_t)((n >> 4) * KT + (k >> 5)) * 512 + frag_slot(n & 15, k & 31);
}

__device__ __forceinline__ v16bf ldfrag(const unsigned short* p, int lane) {
  v8u u = *(const v8u*)(p + (size_t)lane * 16);   // 32B/lane, 1KB contiguous frag
  return __builtin_bit_cast(v16bf, u);
}
__device__ __forceinline__ v8f wmma_bf16(v16bf a, v16bf b, v8f c) {
  return __builtin_amdgcn_wmma_f32_16x16x32_bf16(false, a, false, b, (short)0, c, false, false);
}

// ---------------------------------------------------------------------------
// Workspace layout (bytes), all offsets 256B aligned.
// Zeroed-per-launch region first (A buffers, f32 h, barrier counter).
// ---------------------------------------------------------------------------
static constexpr size_t OFF_A0   = 0;          // 2 * 2mt * 16kt * 512 * 2B = 65536
static constexpr size_t OFF_A1   = 32768;      // 2 * 2mt * 32kt * 512 * 2B = 131072
static constexpr size_t OFF_A2   = 98304;      // 131072
static constexpr size_t OFF_H    = 163840;     // 3*32*512*4 = 196608
static constexpr size_t OFF_BAR  = 360448;     // 256
static constexpr size_t ZERO_SZ  = 360704;
static constexpr size_t OFF_W0   = 360704;     // 96nt*16kt*512*2 = 1572864
static constexpr size_t OFF_W1   = 1933568;    // 128nt*32kt*512*2 = 4194304
static constexpr size_t OFF_W2   = 6127872;    // 4194304
static constexpr size_t OFF_K0P  = 10322176;   // 1572864
static constexpr size_t OFF_WP   = 11895040;   // 32nt*3kt*512*2 = 98304
static constexpr size_t OFF_RB0  = 11993344;   // 1536*4
static constexpr size_t OFF_PB1  = 11999488;   // 2048*4
static constexpr size_t OFF_PB2  = 12007680;   // 2048*4
static constexpr size_t OFF_B0P  = 12015872;   // 1536*4
static constexpr size_t OFF_XP   = 12022016;   // 2000mt*16kt*512*2 = 32768000
static constexpr size_t OFF_GX0  = 44790016;   // 32000*1536*2 = 98304000
static constexpr size_t WS_NEED  = 143094016;

static constexpr size_t OUT_HT   = 16384000;   // 32*1000*512

// ---------------------------------------------------------------------------
// Weight prep: 5,029,888 elements, one thread each (19648 blocks * 256).
// ---------------------------------------------------------------------------
__global__ void __launch_bounds__(256) kprep(
    const float* __restrict__ projW, const float* __restrict__ kern,
    const float* __restrict__ reck,  const float* __restrict__ bias,
    unsigned short* W0, unsigned short* W1, unsigned short* W2,
    unsigned short* K0p, unsigned short* Wp,
    float* rb0, float* pb1, float* pb2, float* b0p)
{
  size_t i = (size_t)blockIdx.x * 256 + threadIdx.x;
  if (i < 786432) {                       // layer-0 RK (W0) + K (K0p): [512,1536]
    int k = (int)(i / 1536), n = (int)(i % 1536);
    int b = n / 96, c = n % 96, g = c >> 5;
    int src = g * 512 + b * 32 + (c & 31);            // gate-permuted source col
    size_t o = bfrag_off(k, n, 16);
    W0[o]  = f2bf(reck[(size_t)k * 1536 + src]);
    K0p[o] = f2bf(kern[(size_t)k * 1536 + src]);
    if (k == 0) { b0p[n] = bias[src]; rb0[n] = bias[1536 + src]; }
    return;
  }
  i -= 786432;
  if (i < 4194304) {                      // layers 1,2 fused [K;RK] -> [1024,2048]
    int l = 1 + (int)(i >> 21);
    size_t j = i & 2097151;
    int k = (int)(j / 2048), n = (int)(j % 2048);
    int b = n / 128, c = n % 128, g = c >> 5;         // gates: z,r,xh,hh
    int col = b * 32 + (c & 31);
    float v;
    if (k < 512) {
      v = (g == 3) ? 0.0f : kern[((size_t)l * 512 + k) * 1536 + g * 512 + col];
    } else {
      int kr = k - 512;
      int off = (g == 3) ? 1024 : g * 512;
      v = (g == 2) ? 0.0f : reck[((size_t)l * 512 + kr) * 1536 + off + col];
    }
    ((l == 1) ? W1 : W2)[bfrag_off(k, n, 32)] = f2bf(v);
    if (k == 0) {
      const float* bi = bias + (size_t)l * 2 * 1536;
      float bb;
      if      (g == 0) bb = bi[col]        + bi[1536 + col];
      else if (g == 1) bb = bi[512 + col]  + bi[1536 + 512 + col];
      else if (g == 2) bb = bi[1024 + col];
      else             bb = bi[1536 + 1024 + col];
      ((l == 1) ? pb1 : pb2)[n] = bb;
    }
    return;
  }
  i -= 4194304;
  {                                       // proj_W [80->96, 512]
    int k = (int)(i / 512), n = (int)(i % 512);
    float v = (k < 80) ? projW[(size_t)k * 512 + n] : 0.0f;
    Wp[bfrag_off(k, n, 3)] = f2bf(v);
  }
}

// ---------------------------------------------------------------------------
// Projection: one 16-row M-tile per block, 8 waves x 4 N-tiles, K=96 (padded).
// Output stored as bf16 A-fragments for the gx0 GEMM.
// ---------------------------------------------------------------------------
__global__ void __launch_bounds__(256) kproj(
    const float* __restrict__ x, const float* __restrict__ projb,
    const unsigned short* __restrict__ Wp, unsigned short* __restrict__ xp)
{
  __shared__ __align__(32) unsigned short la[1536];   // 3 k-tiles of A frags
  const int tid = threadIdx.x, mt = blockIdx.x;
  for (int i = tid; i < 1536; i += 256) la[i] = 0;
  __syncthreads();
  for (int i = tid; i < 1280; i += 256) {
    int m = i / 80, d = i % 80;
    la[(d >> 5) * 512 + frag_slot(m, d & 31)] = f2bf(x[(size_t)(mt * 16 + m) * 80 + d]);
  }
  __syncthreads();
  const int w = tid >> 5, lane = tid & 31;
  v8f acc[4] = {};
  for (int kt = 0; kt < 3; ++kt) {
    v16bf a = ldfrag(la + kt * 512, lane);
#pragma unroll
    for (int j = 0; j < 4; ++j) {
      v16bf b = ldfrag(Wp + (size_t)((w * 4 + j) * 3 + kt) * 512, lane);
      acc[j] = wmma_bf16(a, b, acc[j]);
    }
  }
  const int mrow = (lane >> 4) * 8, nl = lane & 15;
#pragma unroll
  for (int j = 0; j < 4; ++j) {
    int n = (w * 4 + j) * 16 + nl;
    float bv = projb[n];
#pragma unroll
    for (int r = 0; r < 8; ++r) {
      float yv = ftanh(acc[j][r] + bv);
      xp[(size_t)mt * 8192 + (size_t)(n >> 5) * 512 + frag_slot(mrow + r, n & 31)] = f2bf(yv);
    }
  }
}

// ---------------------------------------------------------------------------
// Bulk gx0 = xp @ K0p + b_in0 : M=32000, K=512, N=1536(packed). grid(1000,8).
// ---------------------------------------------------------------------------
__global__ void __launch_bounds__(256) kgx0(
    const unsigned short* __restrict__ xp, const unsigned short* __restrict__ K0p,
    const float* __restrict__ b0p, unsigned short* __restrict__ gx0)
{
  const int tid = threadIdx.x, w = tid >> 5, lane = tid & 31;
  const int mt  = blockIdx.x * 2 + (w & 1);
  const int ntb = blockIdx.y * 12 + (w >> 1) * 3;
  v8f acc[3] = {};
  const unsigned short* Ab = xp + (size_t)mt * 8192;
  for (int kt = 0; kt < 16; ++kt) {
    v16bf a = ldfrag(Ab + kt * 512, lane);
#pragma unroll
    for (int j = 0; j < 3; ++j) {
      v16bf b = ldfrag(K0p + (size_t)((ntb + j) * 16 + kt) * 512, lane);
      acc[j] = wmma_bf16(a, b, acc[j]);
    }
  }
  const int mrow = (lane >> 4) * 8, nl = lane & 15;
#pragma unroll
  for (int j = 0; j < 3; ++j) {
    int n = (ntb + j) * 16 + nl;
    float bv = b0p[n];
#pragma unroll
    for (int r = 0; r < 8; ++r) {
      size_t row = (size_t)mt * 16 + mrow + r;
      gx0[row * 1536 + n] = f2bf(acc[j][r] + bv);
    }
  }
}

// ---------------------------------------------------------------------------
// Persistent recurrent kernel helpers
// ---------------------------------------------------------------------------
__device__ __forceinline__ void gbar(unsigned* bar, unsigned target) {
  __threadfence();
  __syncthreads();
  if (threadIdx.x == 0) {
    atomicAdd(bar, 1u);
    while (__hip_atomic_load(bar, __ATOMIC_RELAXED, __HIP_MEMORY_SCOPE_AGENT) < target)
      __builtin_amdgcn_s_sleep(2);
  }
  __syncthreads();
  __threadfence();
}

// Layers 1/2: A=[inp(t), h(t-1)] 32x1024, W packed 1024x(16x128 cols).
// wave w: M-tile w>>2, gate g=w&3 (two 16-col N-tiles of that gate).
__device__ __forceinline__ void stageL(
    const unsigned short* __restrict__ Ard, const unsigned short* __restrict__ W,
    const float* __restrict__ pb, float* __restrict__ hl, float* lg,
    unsigned short* AnxtH, unsigned short* AcurNext, float* yout,
    int t, int blk, int tid, int w, int lane, int mrow, int nl)
{
  const int mt = w >> 2, g = w & 3;
  v8f a0 = {}, a1 = {};
  const unsigned short* Af = Ard + (size_t)mt * 16384;
  const unsigned short* B0 = W + (size_t)(blk * 8 + g * 2) * 16384;
  const unsigned short* B1 = B0 + 16384;
  for (int kt = 0; kt < 32; ++kt) {
    v16bf a  = ldfrag(Af + kt * 512, lane);
    v16bf b0 = ldfrag(B0 + kt * 512, lane);
    v16bf b1 = ldfrag(B1 + kt * 512, lane);
    a0 = wmma_bf16(a, b0, a0);
    a1 = wmma_bf16(a, b1, a1);
  }
  const int c0 = g * 32 + nl;
  const float bA = pb[blk * 128 + c0];
  const float bB = pb[blk * 128 + c0 + 16];
#pragma unroll
  for (int r = 0; r < 8; ++r) {
    int row = mt * 16 + mrow + r;
    lg[row * 128 + c0]      = a0[r] + bA;
    lg[row * 128 + c0 + 16] = a1[r] + bB;
  }
  __syncthreads();
  for (int it = tid; it < 1024; it += 256) {
    int m = it >> 5, c = it & 31;
    float z    = sigm(lg[m * 128 + c]);
    float rr   = sigm(lg[m * 128 + 32 + c]);
    float cand = ftanh(lg[m * 128 + 64 + c] + rr * lg[m * 128 + 96 + c]);
    int n = blk * 32 + c;
    float hold = hl[m * 512 + n];
    float hnew = z * hold + (1.0f - z) * cand;
    hl[m * 512 + n] = hnew;
    unsigned short us = f2bf(hnew);
    AnxtH[afrag_off(m, 512 + n, 32)] = us;          // h-slot for next step
    if (AcurNext) AcurNext[afrag_off(m, n, 32)] = us;  // inp-slot of next layer
    if (yout) yout[((size_t)m * 1000 + t) * 512 + n] = hnew;
  }
}

__global__ void __launch_bounds__(256) krecur(
    const unsigned short* __restrict__ gx0,
    const unsigned short* __restrict__ W0, const unsigned short* __restrict__ W1,
    const unsigned short* __restrict__ W2,
    const float* __restrict__ rb0, const float* __restrict__ pb1,
    const float* __restrict__ pb2,
    unsigned short* A0, unsigned short* A1, unsigned short* A2,
    float* hbuf, unsigned* bar, float* out)
{
  __shared__ __align__(32) float lg[32 * 128];
  const int tid = threadIdx.x, w = tid >> 5, lane = tid & 31, blk = blockIdx.x;
  const int mrow = (lane >> 4) * 8, nl = lane & 15;
  unsigned btar = 0;
  for (int t = 0; t < 1000; ++t) {
    const int cur = t & 1, nxt = cur ^ 1;
    // ---- stage 0: gh0 = h0 @ RK0 (32x512 @ 512x96-slice) ----
    {
      const unsigned short* Ab = A0 + (size_t)cur * 16384;
      for (int pp = w; pp < 12; pp += 8) {
        const int mt = pp & 1, nt = pp >> 1;
        v8f acc = {};
        const unsigned short* Af = Ab + (size_t)mt * 8192;
        const unsigned short* Bf = W0 + (size_t)(blk * 6 + nt) * 8192;
        for (int kt = 0; kt < 16; ++kt) {
          v16bf a = ldfrag(Af + kt * 512, lane);
          v16bf b = ldfrag(Bf + kt * 512, lane);
          acc = wmma_bf16(a, b, acc);
        }
        const int cl = nt * 16 + nl;
        const float bv = rb0[blk * 96 + cl];
#pragma unroll
        for (int r = 0; r < 8; ++r)
          lg[(mt * 16 + mrow + r) * 96 + cl] = acc[r] + bv;
      }
    }
    __syncthreads();
    for (int it = tid; it < 1024; it += 256) {
      int m = it >> 5, c = it & 31;
      float gz = lg[m * 96 + c], gr = lg[m * 96 + 32 + c], gh = lg[m * 96 + 64 + c];
      size_t gxb = ((size_t)m * 1000 + t) * 1536 + (size_t)blk * 96;
      float z    = sigm(bf2f(gx0[gxb + c]) + gz);
      float rr   = sigm(bf2f(gx0[gxb + 32 + c]) + gr);
      float cand = ftanh(bf2f(gx0[gxb + 64 + c]) + rr * gh);
      int n = blk * 32 + c;
      float hold = hbuf[m * 512 + n];
      float hnew = z * hold + (1.0f - z) * cand;
      hbuf[m * 512 + n] = hnew;
      unsigned short us = f2bf(hnew);
      A0[(size_t)nxt * 16384 + afrag_off(m, n, 16)] = us;   // h0 for t+1
      A1[(size_t)cur * 32768 + afrag_off(m, n, 32)] = us;   // layer-1 input, step t
    }
    btar += NBLK; gbar(bar, btar);
    // ---- stage 1 ----
    stageL(A1 + (size_t)cur * 32768, W1, pb1, hbuf + 16384, lg,
           A1 + (size_t)nxt * 32768, A2 + (size_t)cur * 32768, nullptr,
           t, blk, tid, w, lane, mrow, nl);
    btar += NBLK; gbar(bar, btar);
    // ---- stage 2 (writes outputs) ----
    stageL(A2 + (size_t)cur * 32768, W2, pb2, hbuf + 32768, lg,
           A2 + (size_t)nxt * 32768, nullptr, out,
           t, blk, tid, w, lane, mrow, nl);
    __syncthreads();
  }
  // final hidden states hT[l][b][h]
  for (int it = tid; it < 3072; it += 256) {
    int l = it >> 10, rem = it & 1023, m = rem >> 5, c = rem & 31;
    int n = blk * 32 + c;
    out[OUT_HT + ((size_t)l * 32 + m) * 512 + n] = hbuf[(size_t)l * 16384 + m * 512 + n];
  }
}

// ---------------------------------------------------------------------------
extern "C" void kernel_launch(void* const* d_in, const int* in_sizes, int n_in,
                              void* d_out, int out_size, void* d_ws, size_t ws_size,
                              hipStream_t stream)
{
  (void)in_sizes; (void)n_in; (void)out_size;
  const float* x     = (const float*)d_in[0];
  const float* projW = (const float*)d_in[1];
  const float* projb = (const float*)d_in[2];
  const float* kern  = (const float*)d_in[3];
  const float* reck  = (const float*)d_in[4];
  const float* bias  = (const float*)d_in[5];
  char* ws = (char*)d_ws;
  if (ws_size < WS_NEED) return;   // insufficient scratch (compile-time sized)

  unsigned short* A0  = (unsigned short*)(ws + OFF_A0);
  unsigned short* A1  = (unsigned short*)(ws + OFF_A1);
  unsigned short* A2  = (unsigned short*)(ws + OFF_A2);
  float*          hb  = (float*)(ws + OFF_H);
  unsigned*       bar = (unsigned*)(ws + OFF_BAR);
  unsigned short* W0  = (unsigned short*)(ws + OFF_W0);
  unsigned short* W1  = (unsigned short*)(ws + OFF_W1);
  unsigned short* W2  = (unsigned short*)(ws + OFF_W2);
  unsigned short* K0p = (unsigned short*)(ws + OFF_K0P);
  unsigned short* Wp  = (unsigned short*)(ws + OFF_WP);
  float*          rb0 = (float*)(ws + OFF_RB0);
  float*          pb1 = (float*)(ws + OFF_PB1);
  float*          pb2 = (float*)(ws + OFF_PB2);
  float*          b0p = (float*)(ws + OFF_B0P);
  unsigned short* xp  = (unsigned short*)(ws + OFF_XP);
  unsigned short* gx0 = (unsigned short*)(ws + OFF_GX0);

  hipMemsetAsync(ws, 0, ZERO_SZ, stream);   // h(-1)=0, barrier=0 (graph-safe)

  kprep<<<19648, 256, 0, stream>>>(projW, kern, reck, bias,
                                   W0, W1, W2, K0p, Wp, rb0, pb1, pb2, b0p);
  kproj<<<2000, 256, 0, stream>>>(x, projb, Wp, xp);
  kgx0<<<dim3(1000, 8), 256, 0, stream>>>(xp, K0p, b0p, gx0);
  krecur<<<NBLK, 256, 0, stream>>>(gx0, W0, W1, W2, rb0, pb1, pb2,
                                   A0, A1, A2, hb, bar, (float*)d_out);
}
